// GraphConvLSTM_87608742904454
// MI455X (gfx1250) — compile-verified, matrix operand order checked by reading, not run on hardware
//
#include <hip/hip_runtime.h>
#include <hip/hip_bf16.h>

typedef __bf16 bf16_t;
typedef __attribute__((ext_vector_type(16))) __bf16 v16bf;
typedef __attribute__((ext_vector_type(8)))  __bf16 v8bf;
typedef __attribute__((ext_vector_type(8)))  float   v8f;
typedef unsigned int u32x4 __attribute__((ext_vector_type(4)));
typedef int          i32x4 __attribute__((ext_vector_type(4)));
typedef int          i32x8 __attribute__((ext_vector_type(8)));

#define BATCH   512
#define INDIM   256
#define HDIM    128
#define TSTEPS  100
#define NNODES  22
#define ROWS    176            // 8 batch elements * 22 nodes per workgroup
#define LDSTR   36             // LDS row stride in floats (8*36 % 64 == 32)
#define KQ      32             // channels per workgroup (H split 4 ways)

// workspace byte offsets (all 256B aligned)
#define OFF_A    0u            // 22*22 f32 normalized adjacency
#define OFF_WT   2048u         // 128*128 bf16, Wt[n][j] = gcn_w[j][n]
#define OFF_XG   34816u        // 4*512*128 f32 gate pre-activations
#define OFF_H0   1083392u      // (B*N)*H bf16 state ping
#define OFF_H1   3966976u      // (B*N)*H bf16 state pong
#define OFF_C    6850560u      // (B*N)*H f32 cell state
#define WS_END   12617728u

#if defined(__has_builtin)
#  if __has_builtin(__builtin_amdgcn_tensor_load_to_lds) && \
      __has_builtin(__builtin_amdgcn_s_wait_tensorcnt)
#    define USE_TDM 1
#  endif
#endif
#ifndef USE_TDM
#  define USE_TDM 0
#endif

__device__ __forceinline__ v16bf load_frag(const bf16_t* __restrict__ base, int koff) {
    // base points at row start + ks*32; koff = 0 (lanes 0-15) or 8 (lanes 16-31)
    v8bf lo = *(const v8bf*)(base + koff);
    v8bf hi = *(const v8bf*)(base + koff + 16);
    return __builtin_shufflevector(lo, hi, 0,1,2,3,4,5,6,7,8,9,10,11,12,13,14,15);
}

__device__ __forceinline__ float sigmoidf_fast(float x) {
    return 1.0f / (1.0f + __expf(-x));
}

// ---------------------------------------------------------------------------
// Init: zero h0/h1/c, build Wt (bf16 transpose of gcn_w), build normalized A.
// ---------------------------------------------------------------------------
__global__ __launch_bounds__(256) void init_kernel(
    const float* __restrict__ gcn_w, float* __restrict__ Aout,
    bf16_t* __restrict__ Wt, unsigned int* __restrict__ zbase, long zwords)
{
    long tid = (long)blockIdx.x * 256 + threadIdx.x;
    long stride = (long)gridDim.x * 256;
    for (long i = tid; i < zwords; i += stride) zbase[i] = 0u;
    for (long e = tid; e < HDIM * HDIM; e += stride) {
        int n = (int)(e >> 7), j = (int)(e & 127);
        Wt[e] = (bf16_t)gcn_w[j * HDIM + n];
    }
    if (blockIdx.x == 0 && threadIdx.x == 0) {
        const int cl[5][6] = {{0,2,5,8,11,-1},{0,1,4,7,10,-1},{0,3,6,9,12,15},
                              {9,14,17,19,21,-1},{9,13,16,18,20,-1}};
        float A[NNODES * NNODES];
        for (int i = 0; i < NNODES * NNODES; ++i) A[i] = 0.0f;
        for (int c = 0; c < 5; ++c)
            for (int i = 0; i < 6; ++i) {
                int u = cl[c][i]; if (u < 0) continue;
                for (int j = i + 1; j < 6; ++j) {
                    int v = cl[c][j]; if (v < 0) continue;
                    A[u * NNODES + v] = 1.0f; A[v * NNODES + u] = 1.0f;
                }
            }
        float dinv[NNODES];
        for (int i = 0; i < NNODES; ++i) {
            float d = 0.0f;
            for (int j = 0; j < NNODES; ++j) d += A[i * NNODES + j];
            if (d < 1.0f) d = 1.0f;
            dinv[i] = 1.0f / sqrtf(d);
        }
        for (int i = 0; i < NNODES; ++i)
            for (int j = 0; j < NNODES; ++j)
                Aout[i * NNODES + j] = dinv[i] * A[i * NNODES + j] * dinv[j];
    }
}

// ---------------------------------------------------------------------------
// Gate pre-activations, full fp32 (computed once, reused 100x):
//   XG[g][b][k] = bias_g[k] + sum_j x[b][j] * W_g[k][j]
// ---------------------------------------------------------------------------
__global__ __launch_bounds__(256) void gates_kernel(
    const float* __restrict__ x,
    const float* __restrict__ wi_w, const float* __restrict__ wi_b,
    const float* __restrict__ wf_w, const float* __restrict__ wf_b,
    const float* __restrict__ wo_w, const float* __restrict__ wo_b,
    const float* __restrict__ wc_w, const float* __restrict__ wc_b,
    float* __restrict__ XG)
{
    int gtid = blockIdx.x * 256 + threadIdx.x;   // 4*512*128 threads
    int k = gtid & (HDIM - 1);
    int b = (gtid >> 7) & (BATCH - 1);
    int g = gtid >> 16;
    const float* W; const float* bb;
    if      (g == 0) { W = wi_w; bb = wi_b; }
    else if (g == 1) { W = wf_w; bb = wf_b; }
    else if (g == 2) { W = wo_w; bb = wo_b; }
    else             { W = wc_w; bb = wc_b; }
    const float4* xr = (const float4*)(x + b * INDIM);
    const float4* wr = (const float4*)(W + k * INDIM);
    float acc = bb[k];
#pragma unroll 8
    for (int j = 0; j < INDIM / 4; ++j) {
        float4 xv = xr[j], wv = wr[j];
        acc = fmaf(xv.x, wv.x, acc); acc = fmaf(xv.y, wv.y, acc);
        acc = fmaf(xv.z, wv.z, acc); acc = fmaf(xv.w, wv.w, acc);
    }
    XG[gtid] = acc;
}

// ---------------------------------------------------------------------------
// One recurrence step, fused:
//   TDM: async-stage this WG's XG block (4 x 8 x 32 f32) into LDS
//   Phase A (WMMA): Y = Hin @ W  -> LDS (176 x 32 channel block)
//   Phase B: g = bias + sum_n A[m][n] Y[b,n,k]; LSTM pointwise
// Grid: 256 WGs = 64 batch-blocks x 4 channel-quarters; 256 threads (8 waves).
// ---------------------------------------------------------------------------
__global__ __launch_bounds__(256) void step_kernel(
    const float*  __restrict__ Anorm,   // 22x22
    const bf16_t* __restrict__ Wt,      // [n][j] bf16
    const float*  __restrict__ XG,      // 4 x B x H
    const bf16_t* __restrict__ Hin,     // (B*22) x 128 bf16
    bf16_t*       __restrict__ Hout,    // (B*22) x 128 bf16
    float*        __restrict__ Cst,     // (B*22) x 128 f32
    const float*  __restrict__ gcn_b,   // 128
    float*        __restrict__ outp,    // B x (T*22*128)
    int t)
{
    __shared__ float Yl[ROWS * LDSTR];          // ~25 KB
    __shared__ float XGl[4 * 8 * KQ];           // 4 KB: [gate][b_local][k_local]
    __shared__ float Al[NNODES * NNODES];

    const int tid  = threadIdx.x;
    const int wg   = blockIdx.x;
    const int bblk = wg >> 2;               // batch block (8 elems)
    const int kq   = wg & 3;                // which 32-channel quarter

    // ---- Stage XG block into LDS (async TDM; overlaps the GEMM) ----
#if USE_TDM
    if (tid < 32) {
        unsigned long long ga =
            (unsigned long long)(size_t)(XG + (size_t)bblk * 8 * HDIM + kq * KQ);
        unsigned int lds_off = (unsigned int)(size_t)&XGl[0];
        u32x4 g0 = { 1u,                                  // count=1, user desc
                     lds_off,                             // lds_addr
                     (unsigned int)(ga & 0xFFFFFFFFu),
                     (unsigned int)((ga >> 32) & 0x01FFFFFFu) | 0x80000000u }; // type=2
        i32x8 g1;
        g1[0] = 0x00020000;                 // data_size = 4B
        g1[1] = (int)(32u << 16);           // tensor_dim0 = 32
        g1[2] = (int)(8u << 16);            // tensor_dim1 = 8
        g1[3] = (int)(32u << 16);           // tile_dim0 = 32
        g1[4] = (int)(8u | (4u << 16));     // tile_dim1 = 8, tile_dim2 = 4
        g1[5] = 128;                        // tensor_dim0_stride = 128 (b stride)
        g1[6] = 0;                          // dim0_stride hi | dim1_stride lo16 (=0)
        g1[7] = 1;                          // dim1_stride[47:16]: 65536 (gate stride)
        i32x4 g2 = { 4, 0, 0, 0 };          // tensor_dim2 = 4
        i32x4 g3 = { 0, 0, 0, 0 };
#if __clang_major__ >= 23
        __builtin_amdgcn_tensor_load_to_lds(g0, g1, g2, g3, (i32x8)(0), 0);
#else
        __builtin_amdgcn_tensor_load_to_lds(g0, g1, g2, g3, 0);
#endif
    }
#else
    for (int i = tid; i < 4 * 8 * KQ; i += 256) {
        int g = i >> 8, rem = i & 255;
        int bl = rem >> 5, kl2 = rem & 31;
        XGl[i] = XG[g * BATCH * HDIM + (bblk * 8 + bl) * HDIM + kq * KQ + kl2];
    }
#endif

    for (int i = tid; i < NNODES * NNODES; i += 256) Al[i] = Anorm[i];

    const int lane = tid & 31;
    const int wave = tid >> 5;
    const int l15  = lane & 15;
    const int koff = (lane >> 4) * 8;       // ISA 16-bit A/B lane swizzle

    // ---- Phase A: WMMA GEMM into LDS ----
    const int ntile = wave & 1;             // 2 n-tiles of 16 cols
    const int ncol  = kq * KQ + ntile * 16 + l15;
    const bf16_t* wrow = Wt + ncol * HDIM;
    v16bf bfrag[4];
#pragma unroll
    for (int ks = 0; ks < 4; ++ks) bfrag[ks] = load_frag(wrow + ks * 32, koff);

    const int growb = bblk * ROWS;
    for (int mt = (wave >> 1); mt < 11; mt += 4) {
        v8f acc = {};
        const bf16_t* arow = Hin + (size_t)(growb + mt * 16 + l15) * HDIM;
#pragma unroll
        for (int ks = 0; ks < 4; ++ks) {
            v16bf afr = load_frag(arow + ks * 32, koff);
            acc = __builtin_amdgcn_wmma_f32_16x16x32_bf16(
                false, afr, false, bfrag[ks], (short)0, acc, false, false);
        }
        const int col   = ntile * 16 + l15;
        const int rbase = mt * 16 + (lane >> 4) * 8;
#pragma unroll
        for (int r = 0; r < 8; ++r) Yl[(rbase + r) * LDSTR + col] = acc[r];
    }

#if USE_TDM
    __builtin_amdgcn_s_wait_tensorcnt(0);
#endif
    __syncthreads();

    // ---- Phase B: node mix + LSTM pointwise ----
    // Each wave owns one batch element: rows r = wave*22 + i, channel = lane.
    const int kl = lane;
    const int kg = kq * KQ + kl;
    const int b  = bblk * 8 + wave;
    const float bias = gcn_b[kg];
    const float* xg0 = &XGl[wave * KQ + kl];
#pragma unroll 2
    for (int i = 0; i < NNODES; ++i) {
        const int r = wave * NNODES + i;

        float g = bias;
        const float* av = &Al[i * NNODES];
        const float* yv = &Yl[(wave * NNODES) * LDSTR + kl];
#pragma unroll
        for (int n = 0; n < NNODES; ++n) g = fmaf(av[n], yv[n * LDSTR], g);

        const float i_t  = sigmoidf_fast(xg0[0 * 8 * KQ] + g);
        const float f_t  = sigmoidf_fast(xg0[1 * 8 * KQ] + g);
        const float o_t  = sigmoidf_fast(xg0[2 * 8 * KQ] + g);
        const float ctil = tanhf(xg0[3 * 8 * KQ] + g);

        const size_t gr = (size_t)(growb + r) * HDIM + kg;
        const float c_new = f_t * Cst[gr] + i_t * ctil;
        const float h_new = o_t * tanhf(c_new);
        Cst[gr]  = c_new;
        Hout[gr] = (bf16_t)h_new;
        outp[(size_t)b * (TSTEPS * NNODES * HDIM) +
             (size_t)t * (NNODES * HDIM) + i * HDIM + kg] = h_new;
    }
}

extern "C" void kernel_launch(void* const* d_in, const int* in_sizes, int n_in,
                              void* d_out, int out_size, void* d_ws, size_t ws_size,
                              hipStream_t stream)
{
    const float* x     = (const float*)d_in[0];
    const float* wi_w  = (const float*)d_in[1];
    const float* wi_b  = (const float*)d_in[2];
    const float* wf_w  = (const float*)d_in[3];
    const float* wf_b  = (const float*)d_in[4];
    const float* wo_w  = (const float*)d_in[5];
    const float* wo_b  = (const float*)d_in[6];
    const float* wc_w  = (const float*)d_in[7];
    const float* wc_b  = (const float*)d_in[8];
    const float* gcn_w = (const float*)d_in[9];
    const float* gcn_b = (const float*)d_in[10];

    char* ws = (char*)d_ws;
    float*  A  = (float*)(ws + OFF_A);
    bf16_t* Wt = (bf16_t*)(ws + OFF_WT);
    float*  XG = (float*)(ws + OFF_XG);
    bf16_t* H0 = (bf16_t*)(ws + OFF_H0);
    bf16_t* H1 = (bf16_t*)(ws + OFF_H1);
    float*  C  = (float*)(ws + OFF_C);
    float* outp = (float*)d_out;

    const long zwords = (long)(WS_END - OFF_H0) / 4;   // zero h0,h1,c (contiguous)
    init_kernel<<<2048, 256, 0, stream>>>(gcn_w, A, Wt,
                                          (unsigned int*)(ws + OFF_H0), zwords);
    gates_kernel<<<1024, 256, 0, stream>>>(x, wi_w, wi_b, wf_w, wf_b,
                                           wo_w, wo_b, wc_w, wc_b, XG);
    for (int t = 0; t < TSTEPS; ++t) {
        const bf16_t* Hi = (t & 1) ? H1 : H0;
        bf16_t*       Ho = (t & 1) ? H0 : H1;
        step_kernel<<<256, 256, 0, stream>>>(A, Wt, XG, Hi, Ho, C, gcn_b, outp, t);
    }
}
